// GGCN_41549513621972
// MI455X (gfx1250) — compile-verified, matrix operand order checked by reading, not run on hardware
//
#include <hip/hip_runtime.h>
#include <hip/hip_bf16.h>
#include <cstdint>

typedef __attribute__((ext_vector_type(16))) __bf16 v16bf;
typedef __attribute__((ext_vector_type(2)))  __bf16 v2bf;
typedef __attribute__((ext_vector_type(8)))  float  v8f;
typedef __attribute__((ext_vector_type(2)))  float  v2f;

#define NROW 8192
#define NHID 256
#define NCLS 16
#define NFEA 1024

// ---------- helpers ----------

__device__ __forceinline__ int kb_of(int v, int lane) {
  int base = (lane & 16) ? 8 : 0;
  return (v < 4) ? (base + 2 * v) : (16 + base + 2 * (v - 4));
}

// 16-bit A/B fragment (16x32 / 32x16) per documented wave32 layout.
// Caller passes pointer to (row-selected) data; pairs (k,k+1) are contiguous.
__device__ __forceinline__ v16bf frag_from_bf16(const __bf16* rowp, int lane) {
  v16bf f;
#pragma unroll
  for (int v = 0; v < 8; ++v) {
    int kb = kb_of(v, lane);
    v2bf t = *(const v2bf*)(rowp + kb);
    f[2 * v] = t.x;
    f[2 * v + 1] = t.y;
  }
  return f;
}

__device__ __forceinline__ v16bf frag_from_f32(const float* rowp, int lane) {
  v16bf f;
#pragma unroll
  for (int v = 0; v < 8; ++v) {
    int kb = kb_of(v, lane);
    v2f t = *(const v2f*)(rowp + kb);
    f[2 * v] = (__bf16)t.x;
    f[2 * v + 1] = (__bf16)t.y;
  }
  return f;
}

__device__ __forceinline__ v8f wmma_bf16(v16bf a, v16bf b, v8f c) {
  return __builtin_amdgcn_wmma_f32_16x16x32_bf16(false, a, false, b, (short)0, c,
                                                 false, false);
}

__device__ __forceinline__ float softplus_f(float x) {
  return (x > 20.f) ? x : log1pf(__expf(x));
}
__device__ __forceinline__ float elu_f(float x) {
  return (x > 0.f) ? x : (__expf(x) - 1.f);
}

// ---------- small kernels ----------

__global__ void k_scalars(float* par, float* out_calib,
                          const float* dc0, const float* c0, const float* s0,
                          const float* dc1, const float* c1, const float* s1,
                          const float* dc2, const float* c2, const float* s2) {
  if (threadIdx.x != 0 || blockIdx.x != 0) return;
  const float* dcs[3] = {dc0, dc1, dc2};
  const float* cs[3]  = {c0, c1, c2};
  const float* ss[3]  = {s0, s1, s2};
  const float alphas[3] = {1.0f, 0.0327898228f /* log(0.9/27+1) */, 0.0f};
  for (int l = 0; l < 3; ++l) {
    float a = cs[l][0], b = cs[l][1], c = cs[l][2];
    float m = fmaxf(a, fmaxf(b, c));
    float ea = __expf(a - m), eb = __expf(b - m), ec = __expf(c - m);
    float inv = 1.f / (ea + eb + ec);
    par[l * 8 + 0] = dcs[l][0];
    par[l * 8 + 1] = dcs[l][1];
    par[l * 8 + 2] = ea * inv;
    par[l * 8 + 3] = eb * inv;
    par[l * 8 + 4] = ec * inv;
    par[l * 8 + 5] = softplus_f(ss[l][0]);
    par[l * 8 + 6] = alphas[l];
    par[l * 8 + 7] = 0.f;
  }
  *out_calib = 0.f;
}

__global__ void k_cvt_bf16(const float* __restrict__ src, __bf16* __restrict__ dst, int n) {
  int i = blockIdx.x * blockDim.x + threadIdx.x;
  if (i < n) dst[i] = (__bf16)src[i];
}

__global__ void k_zero_bf16(__bf16* dst, int n) {
  int i = blockIdx.x * blockDim.x + threadIdx.x;
  if (i < n) dst[i] = (__bf16)0.f;
}

// row sum-of-squares over bf16 [NROW][dpad]
__global__ void k_rowsq(const __bf16* __restrict__ wh, int dpad, float* __restrict__ sq) {
  int lane = threadIdx.x & 31;
  int wv = threadIdx.x >> 5;
  int row = blockIdx.x * (blockDim.x >> 5) + wv;
  float s = 0.f;
  for (int c = lane; c < dpad; c += 32) {
    float v = (float)wh[(size_t)row * dpad + c];
    s += v * v;
  }
#pragma unroll
  for (int o = 16; o > 0; o >>= 1) s += __shfl_xor(s, o, 32);
  if (lane == 0) sq[row] = s;
}

// ---------- WMMA GEMM: out = A[M,K](f32) @ W[dout,K](bf16)^T + bias ----------
// one wave per block; wave computes a 16 x (NT*16) output tile
template <int NT>
__global__ void k_gemm_bias(const float* __restrict__ A, const __bf16* __restrict__ W,
                            const float* __restrict__ bias, int K, int dout,
                            float* outf, __bf16* outbf, int obfld, int do_elu) {
  const int lane = threadIdx.x & 31;
  const int i0 = blockIdx.x * 16;
  v8f C[NT];
  const v8f z = {0.f, 0.f, 0.f, 0.f, 0.f, 0.f, 0.f, 0.f};
#pragma unroll
  for (int t = 0; t < NT; ++t) C[t] = z;
  const int arow = i0 + (lane & 15);
  for (int k0 = 0; k0 < K; k0 += 32) {
    v16bf af = frag_from_f32(A + (size_t)arow * K + k0, lane);
#pragma unroll
    for (int t = 0; t < NT; ++t) {
      int wrow = (blockIdx.y * NT + t) * 16 + (lane & 15);
      v16bf bf = frag_from_bf16(W + (size_t)wrow * K + k0, lane);
      C[t] = wmma_bf16(af, bf, C[t]);
    }
  }
  const int hi = (lane >> 4) & 1;
#pragma unroll
  for (int t = 0; t < NT; ++t) {
    int n = (blockIdx.y * NT + t) * 16 + (lane & 15);
    float bv = bias[n];
#pragma unroll
    for (int r = 0; r < 8; ++r) {
      int row = i0 + r + 8 * hi;
      float v = C[t][r] + bv;
      if (do_elu) v = elu_f(v);
      if (outf) outf[(size_t)row * dout + n] = v;
      if (outbf) outbf[(size_t)row * obfld + n] = (__bf16)v;
    }
  }
}

// ---------- fused GGCN layer (flash-attention style) ----------
// D = padded hidden dim (256 for layers 0/1, 32 for layer 2 with dreal=16)
// mode 0/1: prev = alpha*elu(layer_out) + prev (also writes bf16 prev)
// mode 2:   outraw = layer_out (raw logits, first dreal cols)
template <int D>
__global__ void __launch_bounds__(128)
k_fused(const __bf16* __restrict__ whbf, const float* __restrict__ sq,
        const float* __restrict__ adj, const float* __restrict__ par,
        int layer, int dreal, float* previo, __bf16* prevbf, float* outraw,
        int mode) {
  constexpr int NT = D / 16;
  constexpr int KT = D / 32;
  __shared__ __bf16 sWj[32][D + 2];   // Wh_j row-major (B frags for P)
  __shared__ __bf16 sWt[D][34];       // Wh_j transposed (B frags for att@Wh)
  __shared__ __bf16 sAtt[4][16][34];  // per-wave att A staging
  __shared__ float sSq[32];

  const int tid = threadIdx.x;
  const int lane = tid & 31;
  const int wv = tid >> 5;
  const int hi = (lane >> 4) & 1;
  const int i0 = blockIdx.x * 64 + wv * 16;

  const float dca = par[layer * 8 + 0], dcb = par[layer * 8 + 1];
  const float cc0 = par[layer * 8 + 2], cc1 = par[layer * 8 + 3];
  const float cc2 = par[layer * 8 + 4];
  const float sscale = par[layer * 8 + 5], alpha = par[layer * 8 + 6];

  // A fragments of Wh_i (kept in registers for the whole j loop)
  v16bf aW[KT];
  const int arow = i0 + (lane & 15);
#pragma unroll
  for (int kt = 0; kt < KT; ++kt)
    aW[kt] = frag_from_bf16(whbf + (size_t)arow * D + kt * 32, lane);

  float sqi[8], adjd[8];
#pragma unroll
  for (int r = 0; r < 8; ++r) {
    int gi = i0 + r + 8 * hi;
    sqi[r] = sq[gi];
    adjd[r] = adj[(size_t)gi * NROW + gi];
  }

  const v8f z = {0.f, 0.f, 0.f, 0.f, 0.f, 0.f, 0.f, 0.f};
  v8f acc[NT];
#pragma unroll
  for (int t = 0; t < NT; ++t) acc[t] = z;

  for (int jb = 0; jb < NROW; jb += 32) {
    // stage Wh_j rows jb..jb+31 into LDS (row-major + transposed)
    for (int idx = tid; idx < 16 * D; idx += 128) {  // pairs of bf16
      int r = (2 * idx) / D, c = (2 * idx) % D;
      v2bf t = *(const v2bf*)(whbf + (size_t)(jb + r) * D + c);
      *((v2bf*)&sWj[r][c]) = t;
      sWt[c][r] = t.x;
      sWt[c + 1][r] = t.y;
    }
    if (tid < 32) sSq[tid] = sq[jb + tid];
    // prefetch next adj block (speculative; OOB prefetch is dropped)
    __builtin_prefetch(adj + (size_t)(i0 + (tid & 63)) * NROW + ((jb + 32) & (NROW - 1)), 0, 0);
    __syncthreads();

    // P = Wh_i @ Wh_j^T : two 16x16 tiles
    v8f P0 = z, P1 = z;
#pragma unroll
    for (int kt = 0; kt < KT; ++kt) {
      v16bf b0 = frag_from_bf16(&sWj[lane & 15][kt * 32], lane);
      P0 = wmma_bf16(aW[kt], b0, P0);
      v16bf b1 = frag_from_bf16(&sWj[16 + (lane & 15)][kt * 32], lane);
      P1 = wmma_bf16(aW[kt], b1, P1);
    }

    // elementwise gate -> combined signed att (bf16) in LDS
#pragma unroll
    for (int jt = 0; jt < 2; ++jt) {
      int gjc = jb + jt * 16 + (lane & 15);
      float sqj = sSq[jt * 16 + (lane & 15)];
#pragma unroll
      for (int r = 0; r < 8; ++r) {
        int gi = i0 + r + 8 * hi;
        float p = jt ? P1[r] : P0[r];
        float den = fmaxf(sqrtf(sqi[r] * sqj), 1e-9f);
        float e = p / den;
        if (gi == gjc) e = 0.f;
        float a = adj[(size_t)gi * NROW + gjc];
        float deg = adjd[r] / fmaxf(a, 1e-9f) - 1.f;
        float sc = softplus_f(dca * deg + dcb);
        float att = e * a * sc;
        // c0*relu(att) + c1*(-relu(-att)) folded into one matrix
        float attc = cc0 * fmaxf(att, 0.f) - cc1 * fmaxf(-att, 0.f);
        sAtt[wv][r + 8 * hi][jt * 16 + (lane & 15)] = (__bf16)attc;
      }
    }
    asm volatile("s_wait_dscnt 0" ::: "memory");  // intra-wave LDS RAW

    // acc += att(16x32) @ Wh_j(32xD)
    v16bf aA = frag_from_bf16(&sAtt[wv][lane & 15][0], lane);
#pragma unroll
    for (int t = 0; t < NT; ++t) {
      v16bf bf = frag_from_bf16(&sWt[t * 16 + (lane & 15)][0], lane);
      acc[t] = wmma_bf16(aA, bf, acc[t]);
    }
    __syncthreads();
  }

  // epilogue: out = s * (acc + c2 * Wh_i), then residual / raw write
#pragma unroll
  for (int t = 0; t < NT; ++t) {
    int n = t * 16 + (lane & 15);
#pragma unroll
    for (int r = 0; r < 8; ++r) {
      int gi = i0 + r + 8 * hi;
      float wh = (float)whbf[(size_t)gi * D + n];
      float val = sscale * (acc[t][r] + cc2 * wh);
      if (mode == 2) {
        if (n < dreal) outraw[(size_t)gi * dreal + n] = val;
      } else {
        float pv = previo[(size_t)gi * NHID + n];
        float nv = alpha * elu_f(val) + pv;
        previo[(size_t)gi * NHID + n] = nv;
        prevbf[(size_t)gi * NHID + n] = (__bf16)nv;
      }
    }
  }
}

// ---------- final log-softmax / softmax / top-2 calibration ----------
__global__ void k_final(const float* __restrict__ logits, float* __restrict__ out) {
  __shared__ float sred[256];
  int row = blockIdx.x * blockDim.x + threadIdx.x;
  float t1 = 0.f, t2 = 0.f;
  if (row < NROW) {
    float v[NCLS];
    float mx = -1e30f;
#pragma unroll
    for (int c = 0; c < NCLS; ++c) {
      v[c] = logits[(size_t)row * NCLS + c];
      mx = fmaxf(mx, v[c]);
    }
    float s = 0.f;
#pragma unroll
    for (int c = 0; c < NCLS; ++c) s += __expf(v[c] - mx);
    float lse = mx + __logf(s);
#pragma unroll
    for (int c = 0; c < NCLS; ++c) {
      float lp = v[c] - lse;
      float pr = __expf(lp);
      out[(size_t)row * NCLS + c] = lp;
      out[(size_t)NROW * NCLS + 1 + (size_t)row * NCLS + c] = pr;
      if (pr > t1) { t2 = t1; t1 = pr; }
      else if (pr > t2) { t2 = pr; }
    }
  }
  sred[threadIdx.x] = t1 + t2;
  __syncthreads();
  for (int o = 128; o > 0; o >>= 1) {
    if (threadIdx.x < o) sred[threadIdx.x] += sred[threadIdx.x + o];
    __syncthreads();
  }
  if (threadIdx.x == 0) atomicAdd(&out[NROW * NCLS], sred[0] * (1.f / NROW));
}

// ---------- launch ----------
extern "C" void kernel_launch(void* const* d_in, const int* in_sizes, int n_in,
                              void* d_out, int out_size, void* d_ws, size_t ws_size,
                              hipStream_t stream) {
  (void)in_sizes; (void)n_in; (void)out_size; (void)ws_size;
  const float* x    = (const float*)d_in[0];
  const float* adj  = (const float*)d_in[1];
  const float* fcnW = (const float*)d_in[2];
  const float* fcnb = (const float*)d_in[3];
  const float* W0   = (const float*)d_in[4];
  const float* b0   = (const float*)d_in[5];
  const float* dc0  = (const float*)d_in[6];
  const float* c0   = (const float*)d_in[7];
  const float* s0   = (const float*)d_in[8];
  const float* W1   = (const float*)d_in[9];
  const float* b1   = (const float*)d_in[10];
  const float* dc1  = (const float*)d_in[11];
  const float* c1   = (const float*)d_in[12];
  const float* s1   = (const float*)d_in[13];
  const float* W2   = (const float*)d_in[14];
  const float* b2   = (const float*)d_in[15];
  const float* dc2  = (const float*)d_in[16];
  const float* c2   = (const float*)d_in[17];
  const float* s2   = (const float*)d_in[18];
  float* out = (float*)d_out;

  uint8_t* w = (uint8_t*)d_ws;
  auto take = [&](size_t bytes) {
    uint8_t* p = w;
    w += (bytes + 255) & ~size_t(255);
    return p;
  };
  float*  par    = (float*)take(256);
  float*  prevf  = (float*)take((size_t)NROW * NHID * 4);
  __bf16* prevbf = (__bf16*)take((size_t)NROW * NHID * 2);
  __bf16* whbf   = (__bf16*)take((size_t)NROW * NHID * 2);
  float*  sqb    = (float*)take((size_t)NROW * 4);
  float*  logits = (float*)take((size_t)NROW * NCLS * 4);
  __bf16* fWb    = (__bf16*)take((size_t)NHID * NFEA * 2);
  __bf16* W0b    = (__bf16*)take((size_t)NHID * NFEA * 2);
  __bf16* W1b    = (__bf16*)take((size_t)NHID * NHID * 2);
  __bf16* W2b    = (__bf16*)take((size_t)NCLS * NHID * 2);

  k_scalars<<<1, 1, 0, stream>>>(par, out + NROW * NCLS,
                                 dc0, c0, s0, dc1, c1, s1, dc2, c2, s2);
  k_cvt_bf16<<<(NHID * NFEA + 255) / 256, 256, 0, stream>>>(fcnW, fWb, NHID * NFEA);
  k_cvt_bf16<<<(NHID * NFEA + 255) / 256, 256, 0, stream>>>(W0, W0b, NHID * NFEA);
  k_cvt_bf16<<<(NHID * NHID + 255) / 256, 256, 0, stream>>>(W1, W1b, NHID * NHID);
  k_cvt_bf16<<<(NCLS * NHID + 255) / 256, 256, 0, stream>>>(W2, W2b, NCLS * NHID);

  // prev = elu(x @ fcnW.T + fcnb)
  k_gemm_bias<4><<<dim3(NROW / 16, NHID / 64), 32, 0, stream>>>(
      x, fWb, fcnb, NFEA, NHID, prevf, nullptr, NHID, 1);

  // layer 0
  k_gemm_bias<4><<<dim3(NROW / 16, NHID / 64), 32, 0, stream>>>(
      x, W0b, b0, NFEA, NHID, nullptr, whbf, NHID, 0);
  k_rowsq<<<NROW / 8, 256, 0, stream>>>(whbf, NHID, sqb);
  k_fused<NHID><<<NROW / 64, 128, 0, stream>>>(whbf, sqb, adj, par, 0, NHID,
                                               prevf, prevbf, nullptr, 0);
  // layer 1
  k_gemm_bias<4><<<dim3(NROW / 16, NHID / 64), 32, 0, stream>>>(
      prevf, W1b, b1, NHID, NHID, nullptr, whbf, NHID, 0);
  k_rowsq<<<NROW / 8, 256, 0, stream>>>(whbf, NHID, sqb);
  k_fused<NHID><<<NROW / 64, 128, 0, stream>>>(whbf, sqb, adj, par, 1, NHID,
                                               prevf, prevbf, nullptr, 1);
  // layer 2 (dout=16, padded to 32 cols for K=32 WMMA)
  k_zero_bf16<<<(NROW * 32 + 255) / 256, 256, 0, stream>>>(whbf, NROW * 32);
  k_gemm_bias<1><<<dim3(NROW / 16, 1), 32, 0, stream>>>(
      prevf, W2b, b2, NHID, NCLS, nullptr, whbf, 32, 0);
  k_rowsq<<<NROW / 8, 256, 0, stream>>>(whbf, 32, sqb);
  k_fused<32><<<NROW / 64, 128, 0, stream>>>(whbf, sqb, adj, par, 2, NCLS,
                                             nullptr, nullptr, logits, 2);

  k_final<<<NROW / 256, 256, 0, stream>>>(logits, out);
}
